// MambaBlock_17549236372193
// MI455X (gfx1250) — compile-verified, matrix operand order checked by reading, not run on hardware
//
#include <hip/hip_runtime.h>
#include <hip/hip_bf16.h>

// ---------------- problem constants ----------------
#define B_   2
#define T_   1024
#define DM   512            // d_model
#define DI   1024           // d_inner
#define NS   16             // state
#define KC   4              // conv kernel
#define NT   (B_ * T_)      // 2048 tokens

// GEMM tiling
#define BM     128
#define BN     128
#define BK     64           // staged K depth (two 32-deep WMMA steps)
#define LDS_STRIDE 72       // elems per LDS row: 64 data + 8 pad (144B) -> conflict-free b128 reads

// ---------------- vector types ----------------
typedef __attribute__((ext_vector_type(16))) __bf16         v16bf;
typedef __attribute__((ext_vector_type(8)))  float          v8f;
typedef __attribute__((ext_vector_type(8)))  unsigned short ushort8;

static __device__ __forceinline__ unsigned short f2bf(float f) {
    unsigned int u = __float_as_uint(f);
    unsigned int r = u + 0x7FFFu + ((u >> 16) & 1u);   // round-to-nearest-even
    return (unsigned short)(r >> 16);
}

// Global-memory bf16 fragment: two contiguous 16B chunks at p and p+16 elems.
static __device__ __forceinline__ v16bf ldfrag(const unsigned short* p) {
    union U { ushort8 h[2]; v16bf v; } u;
    u.h[0] = *(const ushort8*)(p);
    u.h[1] = *(const ushort8*)(p + 16);
    return u.v;
}

// LDS bf16 fragment via ds_load_b128 x2
static __device__ __forceinline__ v16bf ldfrag_lds(const unsigned short* p) {
    typedef __attribute__((address_space(3))) const ushort8 lds_u8v;
    union U { ushort8 h[2]; v16bf v; } u;
    u.h[0] = *(lds_u8v*)((__attribute__((address_space(3))) const unsigned short*)p);
    u.h[1] = *(lds_u8v*)((__attribute__((address_space(3))) const unsigned short*)(p + 16));
    return u.v;
}

// 16B global -> LDS async copy (ASYNCcnt path), per-lane.
static __device__ __forceinline__ void async_cp16(unsigned short* l, const unsigned short* g) {
    unsigned la = (unsigned)(uintptr_t)(__attribute__((address_space(3))) unsigned short*)l;
    unsigned long long ga = (unsigned long long)(uintptr_t)g;
    asm volatile("global_load_async_to_lds_b128 %0, %1, off" :: "v"(la), "v"(ga) : "memory");
}
#define ASYNC_WAIT_LE8() asm volatile("s_wait_asynccnt 0x8" ::: "memory")
#define ASYNC_WAIT_0()   asm volatile("s_wait_asynccnt 0x0" ::: "memory")

#define WMMA_BF16(a, b, c) \
    __builtin_amdgcn_wmma_f32_16x16x32_bf16(false, (a), false, (b), (short)0, (c), false, false)

static __device__ __forceinline__ float silu_f(float v) {
    return v / (1.0f + __expf(-v));
}

// ---------------- LayerNorm (1 wave / token) + bf16 cast ----------------
__global__ void ln_kernel(const float* __restrict__ x, const float* __restrict__ g,
                          const float* __restrict__ b, unsigned short* __restrict__ xn_bf) {
    int wave = blockIdx.x * (blockDim.x >> 5) + (threadIdx.x >> 5);
    int lane = threadIdx.x & 31;
    if (wave >= NT) return;
    const float* row = x + (size_t)wave * DM;
    float v[DM / 32];
    float s = 0.f, ss = 0.f;
#pragma unroll
    for (int i = 0; i < DM / 32; ++i) {
        float t = row[lane + i * 32];
        v[i] = t; s += t; ss += t * t;
    }
#pragma unroll
    for (int m = 16; m >= 1; m >>= 1) {
        s  += __shfl_xor(s,  m, 32);
        ss += __shfl_xor(ss, m, 32);
    }
    float mu  = s * (1.0f / DM);
    float var = ss * (1.0f / DM) - mu * mu;
    float rs  = rsqrtf(var + 1e-5f);
#pragma unroll
    for (int i = 0; i < DM / 32; ++i) {
        int c = lane + i * 32;
        xn_bf[(size_t)wave * DM + c] = f2bf((v[i] - mu) * rs * g[c] + b[c]);
    }
}

// ---------------- fp32 -> bf16 elementwise ----------------
__global__ void cvt_bf16_kernel(const float* __restrict__ s, unsigned short* __restrict__ d, int n) {
    int i = blockIdx.x * blockDim.x + threadIdx.x;
    if (i < n) d[i] = f2bf(s[i]);
}

// ---------------- LDS-staged WMMA GEMM: 128x128 per workgroup ----------------
// C[m,n] = sum_k A[m,k]*W[n,k] (+R).  A:MxK, W:NxK bf16 row-major, C:MxN fp32.
// 8 waves; wave w computes rows [w*16, w*16+16) x all 128 cols (8 accum tiles).
// A/B 128x64 panels double-buffered in LDS, filled with global_load_async_to_lds_b128.
// B fragments of each 32-deep k-step are preloaded into registers so ds_load
// latency overlaps the WMMA pipe (avoids per-WMMA s_wait_dscnt 0).
__global__ void __launch_bounds__(256)
wmma_gemm_lds(const unsigned short* __restrict__ A, const unsigned short* __restrict__ W,
              float* __restrict__ C, const float* __restrict__ R, int M, int N, int K) {
    __shared__ unsigned short lds[2][2][BM * LDS_STRIDE];   // [buf][A=0/B=1] : 72KB total
    const int tid  = threadIdx.x;
    const int lane = tid & 31;
    const int wave = tid >> 5;
    const int nb   = N / BN;
    const int bm   = blockIdx.x / nb;
    const int bn   = blockIdx.x - bm * nb;
    const unsigned short* Ab = A + (size_t)bm * BM * K;
    const unsigned short* Wb = W + (size_t)bn * BN * K;

    // copy slots: per chunk, 128 rows x 8 segs x 2 matrices of 16B units.
    // thread handles units u = tid + j*256, j=0..3 for each matrix (8 async ops).
    auto issue = [&](int buf, int k0) {
        unsigned short* lA = &lds[buf][0][0];
        unsigned short* lB = &lds[buf][1][0];
#pragma unroll
        for (int j = 0; j < 4; ++j) {
            int u   = tid + j * 256;          // 0..1023
            int row = u >> 3;
            int seg = (u & 7) * 8;            // elem offset within 64-elem row
            async_cp16(lA + row * LDS_STRIDE + seg, Ab + (size_t)row * K + k0 + seg);
            async_cp16(lB + row * LDS_STRIDE + seg, Wb + (size_t)row * K + k0 + seg);
        }
    };

    const int h  = lane >> 4;
    const int mr = lane & 15;
    v8f acc[8];
#pragma unroll
    for (int j = 0; j < 8; ++j) acc[j] = 0.0f;

    const int nChunks = K / BK;
    issue(0, 0);
    int buf = 0;
    for (int kc = 0; kc < nChunks; ++kc) {
        const bool has_next = (kc + 1) < nChunks;
        if (has_next) issue(buf ^ 1, (kc + 1) * BK);
        if (has_next) { ASYNC_WAIT_LE8(); } else { ASYNC_WAIT_0(); }
        __syncthreads();                               // chunk kc visible to all waves

        const unsigned short* lA = &lds[buf][0][0];
        const unsigned short* lB = &lds[buf][1][0];
#pragma unroll
        for (int ks = 0; ks < 2; ++ks) {               // two 32-deep steps per panel
            const int ko = ks * 32 + h * 8;
            v16bf af = ldfrag_lds(lA + (wave * 16 + mr) * LDS_STRIDE + ko);
            v16bf bf[8];
#pragma unroll
            for (int j = 0; j < 8; ++j)                // batch all ds_loads first
                bf[j] = ldfrag_lds(lB + (j * 16 + mr) * LDS_STRIDE + ko);
#pragma unroll
            for (int j = 0; j < 8; ++j)                // then burst the WMMAs
                acc[j] = WMMA_BF16(af, bf[j], acc[j]);
        }

        __syncthreads();                               // done reading before overwrite
        buf ^= 1;
    }

    // epilogue: element r of acc tile j -> C[bm*128 + wave*16 + r + 8h][bn*128 + j*16 + mr]
    if (R) {
#pragma unroll
        for (int j = 0; j < 8; ++j)
#pragma unroll
            for (int r = 0; r < 8; ++r) {
                size_t idx = (size_t)(bm * BM + wave * 16 + r + 8 * h) * N + bn * BN + j * 16 + mr;
                C[idx] = acc[j][r] + R[idx];
            }
    } else {
#pragma unroll
        for (int j = 0; j < 8; ++j)
#pragma unroll
            for (int r = 0; r < 8; ++r) {
                size_t idx = (size_t)(bm * BM + wave * 16 + r + 8 * h) * N + bn * BN + j * 16 + mr;
                C[idx] = acc[j][r];
            }
    }
}

// ---------------- WMMA GEMM, single 16x16 tile per wave (small N = 16) ----
__global__ void wmma_gemm1(const unsigned short* __restrict__ A,
                           const unsigned short* __restrict__ W,
                           float* __restrict__ C, int M, int N, int K) {
    int lane = threadIdx.x & 31;
    int wid  = blockIdx.x * (blockDim.x >> 5) + (threadIdx.x >> 5);
    int tn   = N >> 4;
    int tm   = wid / tn;
    int tnn  = wid - tm * tn;
    if (tm >= (M >> 4)) return;
    int h  = lane >> 4;
    int mr = lane & 15;
    const unsigned short* arow = A + (size_t)(tm * 16 + mr) * K + h * 8;
    const unsigned short* wrow = W + (size_t)(tnn * 16 + mr) * K + h * 8;
    v8f acc = 0.0f;
    for (int k = 0; k < K; k += 32)
        acc = WMMA_BF16(ldfrag(arow + k), ldfrag(wrow + k), acc);
#pragma unroll
    for (int r = 0; r < 8; ++r)
        C[(size_t)(tm * 16 + r + 8 * h) * N + tnn * 16 + mr] = acc[r];
}

// ---------------- depthwise causal conv (K=4) + SiLU ----------------
__global__ void conv_silu_kernel(const float* __restrict__ xz, const float* __restrict__ cw,
                                 const float* __restrict__ cb,
                                 float* __restrict__ xc_f, unsigned short* __restrict__ xc_bf) {
    int i = blockIdx.x * blockDim.x + threadIdx.x;          // over NT*DI
    if (i >= NT * DI) return;
    int row = i / DI, d = i - row * DI;
    int t = row & (T_ - 1);
    int base = row - t;                                     // b*T
    float acc = cb[d];
#pragma unroll
    for (int k = 0; k < KC; ++k) {
        int tt = t - (KC - 1) + k;
        if (tt >= 0) acc += cw[d * KC + k] * xz[(size_t)(base + tt) * (2 * DI) + d];
    }
    float sv = silu_f(acc);
    xc_f[i]  = sv;
    xc_bf[i] = f2bf(sv);
}

// ---------------- fused selective scan (1 thread per (b,d), 16 states in VGPRs) ----
__global__ void scan_kernel(const float* __restrict__ dtp, const float* __restrict__ b_dt,
                            const float* __restrict__ logA, const float* __restrict__ Bt,
                            const float* __restrict__ Ct,  const float* __restrict__ xc,
                            float* __restrict__ y) {
    int gid = blockIdx.x * blockDim.x + threadIdx.x;        // 0 .. B*DI-1
    if (gid >= B_ * DI) return;
    int b = gid / DI, d = gid - b * DI;
    float A[NS];
#pragma unroll
    for (int n = 0; n < NS; ++n) A[n] = -__expf(logA[(size_t)d * NS + n]);
    float bdt = b_dt[d];
    float h[NS];
#pragma unroll
    for (int n = 0; n < NS; ++n) h[n] = 0.f;

    for (int t = 0; t < T_; ++t) {
        int row = b * T_ + t;
        float dv = dtp[(size_t)row * DI + d] + bdt;
        dv = (dv > 20.f) ? dv : log1pf(__expf(dv));         // softplus
        float xv = xc[(size_t)row * DI + d];
        float dx = dv * xv;
        const float* Br = Bt + (size_t)row * NS;
        const float* Cr = Ct + (size_t)row * NS;
        float acc = 0.f;
#pragma unroll
        for (int n = 0; n < NS; ++n) {
            float dA = __expf(A[n] * dv);
            h[n] = dA * h[n] + dx * Br[n];
            acc += h[n] * Cr[n];
        }
        y[(size_t)row * DI + d] = acc;
    }
}

// ---------------- gate: g = y * silu(z), cast bf16 ----------------
__global__ void gate_kernel(const float* __restrict__ xz, const float* __restrict__ y,
                            unsigned short* __restrict__ g_bf) {
    int i = blockIdx.x * blockDim.x + threadIdx.x;
    if (i >= NT * DI) return;
    int row = i / DI, d = i - row * DI;
    float z = xz[(size_t)row * (2 * DI) + DI + d];
    g_bf[i] = f2bf(y[i] * silu_f(z));
}

// ---------------- host side ----------------
static inline size_t alignup(size_t v) { return (v + 255) & ~(size_t)255; }

extern "C" void kernel_launch(void* const* d_in, const int* in_sizes, int n_in,
                              void* d_out, int out_size, void* d_ws, size_t ws_size,
                              hipStream_t stream) {
    const float* x      = (const float*)d_in[0];
    const float* ln_g   = (const float*)d_in[1];
    const float* ln_b   = (const float*)d_in[2];
    const float* W_in   = (const float*)d_in[3];
    const float* conv_w = (const float*)d_in[4];
    const float* conv_b = (const float*)d_in[5];
    const float* W_B    = (const float*)d_in[6];
    const float* W_C    = (const float*)d_in[7];
    const float* W_dt   = (const float*)d_in[8];
    const float* b_dt   = (const float*)d_in[9];
    const float* log_A  = (const float*)d_in[10];
    const float* W_out  = (const float*)d_in[11];
    float* out = (float*)d_out;

    char* ws = (char*)d_ws;
    size_t off = 0;
    auto take = [&](size_t bytes) { char* p = ws + off; off = alignup(off + bytes); return p; };
    unsigned short* xn_bf   = (unsigned short*)take((size_t)NT * DM * 2);
    unsigned short* Win_bf  = (unsigned short*)take((size_t)2 * DI * DM * 2);
    unsigned short* Wdt_bf  = (unsigned short*)take((size_t)DI * DI * 2);
    unsigned short* Wout_bf = (unsigned short*)take((size_t)DM * DI * 2);
    unsigned short* WB_bf   = (unsigned short*)take((size_t)NS * DI * 2);
    unsigned short* WC_bf   = (unsigned short*)take((size_t)NS * DI * 2);
    float*          xz      = (float*)take((size_t)NT * 2 * DI * 4);
    float*          xc_f    = (float*)take((size_t)NT * DI * 4);
    unsigned short* xc_bf   = (unsigned short*)take((size_t)NT * DI * 2);
    float*          dtp     = (float*)take((size_t)NT * DI * 4);
    float*          Bt      = (float*)take((size_t)NT * NS * 4);
    float*          Ct      = (float*)take((size_t)NT * NS * 4);
    float*          yb      = (float*)take((size_t)NT * DI * 4);
    unsigned short* g_bf    = (unsigned short*)take((size_t)NT * DI * 2);
    (void)ws_size; (void)in_sizes; (void)n_in; (void)out_size;

    const int TB = 256;                       // 8 waves / block

    // 1) LayerNorm + bf16 cast (1 wave per token)
    ln_kernel<<<NT / 8, TB, 0, stream>>>(x, ln_g, ln_b, xn_bf);

    // 2) weight casts to bf16
    cvt_bf16_kernel<<<(2 * DI * DM + TB - 1) / TB, TB, 0, stream>>>(W_in,  Win_bf,  2 * DI * DM);
    cvt_bf16_kernel<<<(DI * DI     + TB - 1) / TB, TB, 0, stream>>>(W_dt,  Wdt_bf,  DI * DI);
    cvt_bf16_kernel<<<(DM * DI     + TB - 1) / TB, TB, 0, stream>>>(W_out, Wout_bf, DM * DI);
    cvt_bf16_kernel<<<(NS * DI     + TB - 1) / TB, TB, 0, stream>>>(W_B,   WB_bf,   NS * DI);
    cvt_bf16_kernel<<<(NS * DI     + TB - 1) / TB, TB, 0, stream>>>(W_C,   WC_bf,   NS * DI);

    // 3) xz = xn @ W_in^T   (M=2048, N=2048, K=512) : LDS-staged async WMMA GEMM
    wmma_gemm_lds<<<(NT / BM) * ((2 * DI) / BN), TB, 0, stream>>>(
        xn_bf, Win_bf, xz, nullptr, NT, 2 * DI, DM);

    // 4) depthwise causal conv + SiLU
    conv_silu_kernel<<<(NT * DI) / TB, TB, 0, stream>>>(xz, conv_w, conv_b, xc_f, xc_bf);

    // 5) dt_pre = xc @ W_dt^T (M=2048,N=1024,K=1024); B,C projections (N=16)
    wmma_gemm_lds<<<(NT / BM) * (DI / BN), TB, 0, stream>>>(
        xc_bf, Wdt_bf, dtp, nullptr, NT, DI, DI);
    {
        int waves = (NT / 16) * (NS / 16);
        wmma_gemm1<<<(waves + 7) / 8, TB, 0, stream>>>(xc_bf, WB_bf, Bt, NT, NS, DI);
        wmma_gemm1<<<(waves + 7) / 8, TB, 0, stream>>>(xc_bf, WC_bf, Ct, NT, NS, DI);
    }

    // 6) fused softplus + selective scan (fp32, states in registers)
    scan_kernel<<<(B_ * DI) / TB, TB, 0, stream>>>(dtp, b_dt, log_A, Bt, Ct, xc_f, yb);

    // 7) gate
    gate_kernel<<<(NT * DI) / TB, TB, 0, stream>>>(xz, yb, g_bf);

    // 8) out = residual + g @ W_out^T  (M=2048, N=512, K=1024)
    wmma_gemm_lds<<<(NT / BM) * (DM / BN), TB, 0, stream>>>(
        g_bf, Wout_bf, out, x, NT, DM, DI);
}
